// Transformer_35416300323552
// MI455X (gfx1250) — matrix operand order repaired and checked by reading, not compiled, not run on hardware
//
#include <hip/hip_runtime.h>
#include <hip/hip_bf16.h>
#include <math.h>

// ---------------------------------------------------------------------------
// Types for CDNA5 WMMA (wave32): D(16x16 f32) = A(16x32 bf16) x B(32x16 bf16) + C
// ---------------------------------------------------------------------------
typedef __attribute__((ext_vector_type(16))) __bf16 bf16x16;
typedef __attribute__((ext_vector_type(8)))  float  f32x8;

#define GF_TRANSB 1
#define GF_BIAS   2
#define GF_RELU   4

// ---------------------------------------------------------------------------
// Generic batched GEMM: C = alpha * A @ B (+ bias) (+relu)
//   block = 128 threads (4 waves), tile 64x64, K stepped by 32 through LDS.
//   Wave (wm,wn) in 2x2 grid owns a 32x32 region = 2x2 WMMA 16x16 tiles.
//   Batch z decomposed as (b1 = z / bdiv, b0 = z % bdiv) with separate strides,
//   which covers the (batch, head) indexing of attention GEMMs.
//   A, B are fp32 in global memory, converted to bf16 while staging to LDS.
//   B is staged TRANSPOSED in LDS so B-fragment loads are two b128 per lane.
//   Global->register->LDS software pipeline: tile k+1 global loads are issued
//   before the WMMAs of tile k, so HBM/L2 latency overlaps matrix compute.
// ---------------------------------------------------------------------------
__global__ void __launch_bounds__(128)
gemm_wmma_kernel(const float* __restrict__ A, const float* __restrict__ Bm,
                 float* __restrict__ C, const float* __restrict__ bias,
                 int M, int N, int K, int lda, int ldb, int ldc,
                 long sA1, long sA0, long sB1, long sB0, long sC1, long sC0,
                 int bdiv, float alpha, int flags)
{
    __shared__ __bf16 As[64 * 32];   // A tile, row-major [row][k]
    __shared__ __bf16 Bt[64 * 32];   // B tile transposed [n][k]

    const int z  = blockIdx.z;
    const int b1 = z / bdiv;
    const int b0 = z % bdiv;
    A  += b1 * sA1 + b0 * sA0;
    Bm += b1 * sB1 + b0 * sB0;
    C  += b1 * sC1 + b0 * sC0;

    const int t    = threadIdx.x;
    const int lane = t & 31;
    const int wave = t >> 5;        // 0..3
    const int wm   = wave >> 1;     // wave row 0..1
    const int wn   = wave & 1;      // wave col 0..1
    const int half = lane >> 4;     // 0 | 1 (half-wave)
    const int lm   = lane & 15;

    const int row0 = blockIdx.y * 64;
    const int col0 = blockIdx.x * 64;

    f32x8 acc[2][2] = {};
    const bool tb = (flags & GF_TRANSB) != 0;

    // per-thread staging slots (see index math below)
    const int ur  = (t >> 1) & 63;       // tile row used by the prefetch in fetchA
    float4 ra[4];                        // A tile: 4 x float4 per thread
    float4 rbv[4];                       // B tile (transposed source rows)
    float  rbs[16];                      // B tile (row-major source)

    auto fetchA = [&](int kb) {
        if (kb + 32 < K)
            __builtin_prefetch(A + (long)(row0 + ur) * lda + kb + 32, 0, 1);
#pragma unroll
        for (int i = 0; i < 4; ++i) {
            int u  = t + i * 128;        // 0..511 float4 slots
            int r  = u >> 3;             // row in tile
            int kq = (u & 7) * 4;        // k offset
            ra[i] = *(const float4*)(A + (long)(row0 + r) * lda + kb + kq);
        }
    };
    auto fetchB = [&](int kb) {
        if (tb) {
            // source element (k, n) = Bm[(col0+n)*ldb + kb+k] : rows contiguous
#pragma unroll
            for (int i = 0; i < 4; ++i) {
                int u  = t + i * 128;
                int n  = u >> 3;
                int kq = (u & 7) * 4;
                rbv[i] = *(const float4*)(Bm + (long)(col0 + n) * ldb + kb + kq);
            }
        } else {
            // source element (k, n) = Bm[(kb+k)*ldb + col0+n] : coalesced over n
#pragma unroll
            for (int i = 0; i < 16; ++i) {
                int u = t + i * 128;     // 0..2047
                int n = u & 63;
                int k = u >> 6;
                rbs[i] = Bm[(long)(kb + k) * ldb + col0 + n];
            }
        }
    };
    auto storeTiles = [&]() {
#pragma unroll
        for (int i = 0; i < 4; ++i) {
            int u  = t + i * 128;
            int r  = u >> 3;
            int kq = (u & 7) * 4;
            __bf16* d = &As[r * 32 + kq];
            d[0] = (__bf16)ra[i].x; d[1] = (__bf16)ra[i].y;
            d[2] = (__bf16)ra[i].z; d[3] = (__bf16)ra[i].w;
        }
        if (tb) {
#pragma unroll
            for (int i = 0; i < 4; ++i) {
                int u  = t + i * 128;
                int n  = u >> 3;
                int kq = (u & 7) * 4;
                __bf16* d = &Bt[n * 32 + kq];
                d[0] = (__bf16)rbv[i].x; d[1] = (__bf16)rbv[i].y;
                d[2] = (__bf16)rbv[i].z; d[3] = (__bf16)rbv[i].w;
            }
        } else {
#pragma unroll
            for (int i = 0; i < 16; ++i) {
                int u = t + i * 128;
                int n = u & 63;
                int k = u >> 6;
                Bt[n * 32 + k] = (__bf16)rbs[i];
            }
        }
    };

    // prologue: fetch first tile into registers
    fetchA(0);
    fetchB(0);

    for (int kb = 0; kb < K; kb += 32) {
        __syncthreads();                 // LDS free (previous frags consumed)
        storeTiles();
        __syncthreads();                 // LDS tiles visible

        // issue next tile's global loads before this tile's WMMAs
        if (kb + 32 < K) {
            fetchA(kb + 32);
            fetchB(kb + 32);
        }

        // ---- load fragments (ISA 7.12.2 layouts) ----
        bf16x16 afrag[2], bfrag[2];
#pragma unroll
        for (int ti = 0; ti < 2; ++ti) {
            // A: lane lm = M row; VGPR0-3: K = half*8 + {0..7}; VGPR4-7: +16
            const __bf16* p = &As[(wm * 32 + ti * 16 + lm) * 32 + half * 8];
            ((uint4*)&afrag[ti])[0] = *(const uint4*)(p);
            ((uint4*)&afrag[ti])[1] = *(const uint4*)(p + 16);
        }
#pragma unroll
        for (int tj = 0; tj < 2; ++tj) {
            // B: lane lm = N col; VGPR0-3: K = half*8 + {0..7}; VGPR4-7: +16
            const __bf16* p = &Bt[(wn * 32 + tj * 16 + lm) * 32 + half * 8];
            ((uint4*)&bfrag[tj])[0] = *(const uint4*)(p);
            ((uint4*)&bfrag[tj])[1] = *(const uint4*)(p + 16);
        }

#pragma unroll
        for (int ti = 0; ti < 2; ++ti)
#pragma unroll
            for (int tj = 0; tj < 2; ++tj)
                acc[ti][tj] = __builtin_amdgcn_wmma_f32_16x16x32_bf16(
                    false, afrag[ti], false, bfrag[tj],
                    (short)0, acc[ti][tj], false, false);
    }

    // ---- epilogue: C layout = lane%16 -> N, VGPR r -> M=r (+8 upper half) ----
#pragma unroll
    for (int ti = 0; ti < 2; ++ti) {
#pragma unroll
        for (int tj = 0; tj < 2; ++tj) {
            const int col   = col0 + wn * 32 + tj * 16 + lm;
            const float bv  = (flags & GF_BIAS) ? bias[col] : 0.f;
            const int rbase = row0 + wm * 32 + ti * 16 + half * 8;
#pragma unroll
            for (int r = 0; r < 8; ++r) {
                float v = acc[ti][tj][r] * alpha + bv;
                if (flags & GF_RELU) v = fmaxf(v, 0.f);
                C[(long)(rbase + r) * ldc + col] = v;
            }
        }
    }
}

// ---------------------------------------------------------------------------
// out[row] = LayerNorm(a[row] + r[row]) * g + beta    (D = 512, block = 256)
// ---------------------------------------------------------------------------
__global__ void __launch_bounds__(256)
ln_kernel(const float* __restrict__ a, const float* __restrict__ r,
          const float* __restrict__ g, const float* __restrict__ beta,
          float* __restrict__ out)
{
    __shared__ float red1[256], red2[256];
    const int t = threadIdx.x;
    const long base = (long)blockIdx.x * 512;
    float v0 = a[base + t];
    float v1 = a[base + t + 256];
    if (r) { v0 += r[base + t]; v1 += r[base + t + 256]; }
    red1[t] = v0 + v1;
    red2[t] = v0 * v0 + v1 * v1;
    __syncthreads();
    for (int off = 128; off > 0; off >>= 1) {
        if (t < off) { red1[t] += red1[t + off]; red2[t] += red2[t + off]; }
        __syncthreads();
    }
    const float mu  = red1[0] * (1.f / 512.f);
    const float var = red2[0] * (1.f / 512.f) - mu * mu;
    const float inv = rsqrtf(var + 1e-5f);
    out[base + t]       = (v0 - mu) * inv * g[t]       + beta[t];
    out[base + t + 256] = (v1 - mu) * inv * g[t + 256] + beta[t + 256];
}

// ---------------------------------------------------------------------------
// Row softmax over attention scores (T = 256 = blockDim), optional causal mask.
// One block per (b, h, s) row.
// ---------------------------------------------------------------------------
__global__ void __launch_bounds__(256)
attn_softmax_kernel(float* __restrict__ sc, int S, int causal)
{
    __shared__ float red[256];
    const int t = threadIdx.x;
    const long row = blockIdx.x;
    const int q = (int)(row % S);
    float* p = sc + row * 256;
    const bool ok = (!causal) || (t <= q);
    const float x = ok ? p[t] : -INFINITY;
    red[t] = x; __syncthreads();
    for (int off = 128; off > 0; off >>= 1) {
        if (t < off) red[t] = fmaxf(red[t], red[t + off]);
        __syncthreads();
    }
    const float m = red[0]; __syncthreads();
    const float e = ok ? __expf(x - m) : 0.f;
    red[t] = e; __syncthreads();
    for (int off = 128; off > 0; off >>= 1) {
        if (t < off) red[t] += red[t + off];
        __syncthreads();
    }
    p[t] = e / red[0];
}

// ---------------------------------------------------------------------------
// In-place softmax over long rows (vocab = 32000), one block per row.
// ---------------------------------------------------------------------------
__global__ void __launch_bounds__(256)
softmax_rows_kernel(float* __restrict__ x, int N)
{
    __shared__ float red[256];
    const int t = threadIdx.x;
    float* p = x + (long)blockIdx.x * N;
    float m = -INFINITY;
    for (int i = t; i < N; i += 256) m = fmaxf(m, p[i]);
    red[t] = m; __syncthreads();
    for (int off = 128; off > 0; off >>= 1) {
        if (t < off) red[t] = fmaxf(red[t], red[t + off]);
        __syncthreads();
    }
    m = red[0]; __syncthreads();
    float s = 0.f;
    for (int i = t; i < N; i += 256) s += __expf(p[i] - m);
    red[t] = s; __syncthreads();
    for (int off = 128; off > 0; off >>= 1) {
        if (t < off) red[t] += red[t + off];
        __syncthreads();
    }
    const float inv = 1.f / red[0];
    for (int i = t; i < N; i += 256) p[i] = __expf(p[i] - m) * inv;
}

// ---------------------------------------------------------------------------
// out[b,s,:] = emb[tok[b,s]] + sinusoidal PE(s)   (D = 512, block = 256)
// ---------------------------------------------------------------------------
__global__ void __launch_bounds__(256)
embed_pe_kernel(const int* __restrict__ tok, const float* __restrict__ emb,
                float* __restrict__ out, int L)
{
    const int bs = blockIdx.x;
    const int s = bs % L;
    const int token = tok[bs];
    const int t = threadIdx.x;
#pragma unroll
    for (int j = 0; j < 2; ++j) {
        const int d = t + j * 256;
        const int i = d >> 1;
        const float ang = (float)s * powf(10000.f, -(2.f * (float)i) / 512.f);
        const float pe = (d & 1) ? cosf(ang) : sinf(ang);
        out[(long)bs * 512 + d] = emb[(long)token * 512 + d] + pe;
    }
}

// ---------------------------------------------------------------------------
// wqkv [3, H=8, D=512, DK=64] -> [3, D=512, H*DK=512] so QKV projections become
// single M=2048 GEMMs.  3*512*512 elements, grid = 3072 x 256.
// ---------------------------------------------------------------------------
__global__ void __launch_bounds__(256)
repack_qkv_kernel(const float* __restrict__ w, float* __restrict__ out)
{
    const int idx = blockIdx.x * 256 + threadIdx.x;
    const int m   = idx / (512 * 512);
    const int rem = idx % (512 * 512);
    const int d   = rem / 512;
    const int n   = rem % 512;
    const int h   = n >> 6, k = n & 63;
    out[idx] = w[(((m * 8 + h) * 512) + d) * 64 + k];
}

// ---------------------------------------------------------------------------
// Host orchestration
// ---------------------------------------------------------------------------
extern "C" void kernel_launch(void* const* d_in, const int* in_sizes, int n_in,
                              void* d_out, int out_size, void* d_ws, size_t ws_size,
                              hipStream_t stream)
{
    (void)in_sizes; (void)n_in; (void)out_size; (void)ws_size;
    const int*   x         = (const int*)  d_in[0];
    const int*   y         = (const int*)  d_in[1];
    const float* emb_src   = (const float*)d_in[2];
    const float* emb_tgt   = (const float*)d_in[3];
    const float* enc_wqkv  = (const float*)d_in[4];
    const float* enc_wo    = (const float*)d_in[5];
    const float* enc_ff1w  = (const float*)d_in[6];
    const float* enc_ff1b  = (const float*)d_in[7];
    const float* enc_ff2w  = (const float*)d_in[8];
    const float* enc_ff2b  = (const float*)d_in[9];
    const float* enc_lng   = (const float*)d_in[10];
    const float* enc_lnb   = (const float*)d_in[11];
    const float* dec_wqkv1 = (const float*)d_in[12];
    const float* dec_wo1   = (const float*)d_in[13];
    const float* dec_wqkv2 = (const float*)d_in[14];
    const float* dec_wo2   = (const float*)d_in[15];
    const float* dec_ff1w  = (const float*)d_in[16];
    const float* dec_ff1b  = (const float*)d_in[17];
    const float* dec_ff2w  = (const float*)d_in[18];
    const float* dec_ff2b  = (const float*)d_in[19];
    const float* dec_lng   = (const float*)d_in[20];
    const float* dec_lnb   = (const float*)d_in[21];
    const float* final_w   = (const float*)d_in[22];
    const float* final_b   = (const float*)d_in[23];
    float* out = (float*)d_out;

    constexpr int  Bb = 8, L = 256, D = 512, Hh = 8, DK = 64, NL = 6, DFF = 2048, V = 32000;
    constexpr int  MM = Bb * L;                 // 2048 rows in the folded [B*L, D] view
    constexpr long WQKV_L = 3L * Hh * D * DK;   // per-layer wqkv stride

    // workspace carve-out (all fp32)
    float* ws = (float*)d_ws;
    size_t o = 0;
    auto alloc = [&](size_t n) { float* p = ws + o; o += n; return p; };
    float* he    = alloc((size_t)MM * D);       // encoder stream
    float* hd    = alloc((size_t)MM * D);       // decoder stream
    float* wperm = alloc(3L * D * D);           // permuted qkv weights (per layer)
    float* qa    = alloc((size_t)MM * D);
    float* ka    = alloc((size_t)MM * D);
    float* va    = alloc((size_t)MM * D);
    float* sc    = alloc((size_t)Bb * Hh * L * L);
    float* oa    = alloc((size_t)MM * D);
    float* f1    = alloc((size_t)MM * DFF);
    float* t2    = alloc((size_t)MM * D);

    auto gemm = [&](const float* A, const float* Bm, float* C, const float* bias,
                    int M, int N, int K, int lda, int ldb, int ldc,
                    long sA1, long sA0, long sB1, long sB0, long sC1, long sC0,
                    int nbatch, int bdiv, float alpha, int flags) {
        dim3 g(N / 64, M / 64, nbatch), blk(128);
        gemm_wmma_kernel<<<g, blk, 0, stream>>>(A, Bm, C, bias, M, N, K,
                                                lda, ldb, ldc,
                                                sA1, sA0, sB1, sB0, sC1, sC0,
                                                bdiv, alpha, flags);
    };

    // hio = LN(hio + MHA(qsrc -> kvsrc))
    auto attention = [&](const float* qsrc, const float* kvsrc,
                         const float* wqkv_layer, const float* wo_layer,
                         const float* lg, const float* lb,
                         float* hio, int causal) {
        repack_qkv_kernel<<<3072, 256, 0, stream>>>(wqkv_layer, wperm);
        gemm(qsrc,  wperm + 0L,            qa, nullptr, MM, D, D, D, D, D,
             0, 0, 0, 0, 0, 0, 1, 1, 1.f, 0);
        gemm(kvsrc, wperm + (long)D * D,   ka, nullptr, MM, D, D, D, D, D,
             0, 0, 0, 0, 0, 0, 1, 1, 1.f, 0);
        gemm(kvsrc, wperm + 2L * D * D,    va, nullptr, MM, D, D, D, D, D,
             0, 0, 0, 0, 0, 0, 1, 1, 1.f, 0);
        // scores[b,h] = (Q K^T) / sqrt(DK) : 64-way batched, B transposed
        gemm(qa, ka, sc, nullptr, L, L, DK, D, D, L,
             (long)L * D, DK, (long)L * D, DK, (long)Hh * L * L, (long)L * L,
             Bb * Hh, Hh, 0.125f, GF_TRANSB);
        attn_softmax_kernel<<<Bb * Hh * L, 256, 0, stream>>>(sc, L, causal);
        // O[b,h] = P V
        gemm(sc, va, oa, nullptr, L, DK, L, L, D, D,
             (long)Hh * L * L, (long)L * L, (long)L * D, DK, (long)L * D, DK,
             Bb * Hh, Hh, 1.f, 0);
        // output projection + residual LayerNorm
        gemm(oa, wo_layer, t2, nullptr, MM, D, D, D, D, D,
             0, 0, 0, 0, 0, 0, 1, 1, 1.f, 0);
        ln_kernel<<<MM, 256, 0, stream>>>(hio, t2, lg, lb, hio);
    };

    // hio = LN(hio + FFN(hio))
    auto ffn = [&](const float* w1, const float* b1,
                   const float* w2, const float* b2,
                   const float* lg, const float* lb, float* hio) {
        gemm(hio, w1, f1, b1, MM, DFF, D, D, DFF, DFF,
             0, 0, 0, 0, 0, 0, 1, 1, 1.f, GF_BIAS | GF_RELU);
        gemm(f1, w2, t2, b2, MM, D, DFF, DFF, D, D,
             0, 0, 0, 0, 0, 0, 1, 1, 1.f, GF_BIAS);
        ln_kernel<<<MM, 256, 0, stream>>>(hio, t2, lg, lb, hio);
    };

    // ---- embeddings + positional encodings ----
    embed_pe_kernel<<<MM, 256, 0, stream>>>(x, emb_src, he, L);
    embed_pe_kernel<<<MM, 256, 0, stream>>>(y, emb_tgt, hd, L);

    // ---- encoder stack ----
    for (int i = 0; i < NL; ++i) {
        attention(he, he, enc_wqkv + i * WQKV_L, enc_wo + (long)i * D * D,
                  enc_lng + (size_t)(i * 2 + 0) * D, enc_lnb + (size_t)(i * 2 + 0) * D,
                  he, /*causal=*/0);
        ffn(enc_ff1w + (long)i * D * DFF, enc_ff1b + (long)i * DFF,
            enc_ff2w + (long)i * DFF * D, enc_ff2b + (long)i * D,
            enc_lng + (size_t)(i * 2 + 1) * D, enc_lnb + (size_t)(i * 2 + 1) * D, he);
    }

    // ---- decoder stack ----
    for (int i = 0; i < NL; ++i) {
        attention(hd, hd, dec_wqkv1 + i * WQKV_L, dec_wo1 + (long)i * D * D,
                  dec_lng + (size_t)(i * 3 + 0) * D, dec_lnb + (size_t)(i * 3 + 0) * D,
                  hd, /*causal=*/1);
        attention(hd, he, dec_wqkv2 + i * WQKV_L, dec_wo2 + (long)i * D * D,
                  dec_lng + (size_t)(i * 3 + 1) * D, dec_lnb + (size_t)(i * 3 + 1) * D,
                  hd, /*causal=*/0);
        ffn(dec_ff1w + (long)i * D * DFF, dec_ff1b + (long)i * DFF,
            dec_ff2w + (long)i * DFF * D, dec_ff2b + (long)i * D,
            dec_lng + (size_t)(i * 3 + 2) * D, dec_lnb + (size_t)(i * 3 + 2) * D, hd);
    }

    // ---- final projection (into d_out) + in-place vocab softmax ----
    gemm(hd, final_w, out, final_b, MM, V, D, D, V, V,
         0, 0, 0, 0, 0, 0, 1, 1, 1.f, GF_BIAS);
    softmax_rows_kernel<<<MM, 256, 0, stream>>>(out, V);
}
